// CrossFrameAttention_42666205118801
// MI455X (gfx1250) — matrix operand order repaired and checked by reading, not compiled, not run on hardware
//
#include <hip/hip_runtime.h>
#include <hip/hip_bf16.h>

// Problem constants
#define B_   4
#define N_   2048
#define M_   2048
#define H_   8
#define DH_  64
#define DIM_ 512

typedef __attribute__((ext_vector_type(16))) __bf16         v16bf;
typedef __attribute__((ext_vector_type(8)))  float          v8f;
typedef __attribute__((ext_vector_type(16))) unsigned short v16u16;
typedef __attribute__((ext_vector_type(8)))  unsigned short v8u16;

__device__ __forceinline__ unsigned short f2bf(float f) {
  return __builtin_bit_cast(unsigned short, (__bf16)f);   // native cvt, RNE
}
__device__ __forceinline__ v16bf asbf(v16u16 v) { return __builtin_bit_cast(v16bf, v); }
__device__ __forceinline__ v16u16 cat8(v8u16 lo, v8u16 hi) {
  v16u16 r;
#pragma unroll
  for (int j = 0; j < 8; ++j) { r[j] = lo[j]; r[8 + j] = hi[j]; }
  return r;
}

// Async 16B global -> LDS copy (ASYNCcnt tracked).  VDST = LDS byte address,
// VADDR = 64-bit global address, SADDR = off.
__device__ __forceinline__ void async_b128(unsigned lds_addr, const void* gptr) {
  asm volatile("global_load_async_to_lds_b128 %0, %1, off"
               :: "v"(lds_addr), "v"(gptr) : "memory");
}
__device__ __forceinline__ unsigned lds_off(const void* p) {
  return (unsigned)(unsigned long long)p;   // flat LDS aperture: low 32 bits = LDS offset
}

// ---------------------------------------------------------------------------
// Kernel 1: Y = X(f32)[8192,512] @ W(f32)[512,512] -> bf16, layout per MODE:
//   MODE 0: out[((b*H+h)*2048 + n)*64 + d]   (Q / K  : [B,H,N,64])
//   MODE 1: out[((b*H+h)*64 + d)*2048 + n]   (V^T    : [B,H,64,M])
// grid (64, 8), block 256. Wave computes 16 rows x 64 cols.
// ---------------------------------------------------------------------------
template <int MODE>
__global__ __launch_bounds__(256) void proj_kernel(const float* __restrict__ X,
                                                   const float* __restrict__ W,
                                                   unsigned short* __restrict__ out) {
  __shared__ unsigned short ldsW[64 * 40];   // W tile transposed [n][k], pitch 40
  const int tid  = threadIdx.x;
  const int lane = tid & 31;
  const int wave = tid >> 5;
  const int rbase = blockIdx.x * 128 + wave * 16;
  const int cbase = blockIdx.y * 64;
  const int l16 = lane & 15;
  const int kb8 = (lane >> 4) * 8;

  v8f acc[4];
#pragma unroll
  for (int t = 0; t < 4; ++t) acc[t] = (v8f)0.f;

  const float* xrow = X + (size_t)(rbase + l16) * DIM_;

  for (int kc = 0; kc < DIM_; kc += 32) {
#pragma unroll
    for (int it = 0; it < 8; ++it) {
      int idx = tid + it * 256;
      int k = idx >> 6, n = idx & 63;
      ldsW[n * 40 + k] = f2bf(W[(size_t)(kc + k) * DIM_ + cbase + n]);
    }
    __syncthreads();

    v16u16 a;
#pragma unroll
    for (int j = 0; j < 8; ++j) {
      a[j]     = f2bf(xrow[kc + kb8 + j]);
      a[8 + j] = f2bf(xrow[kc + 16 + kb8 + j]);
    }
    v16bf av = asbf(a);

#pragma unroll
    for (int nt = 0; nt < 4; ++nt) {
      const unsigned short* bp = &ldsW[(nt * 16 + l16) * 40 + kb8];
      v16bf bv = asbf(cat8(*(const v8u16*)bp, *(const v8u16*)(bp + 16)));
      acc[nt] = __builtin_amdgcn_wmma_f32_16x16x32_bf16(false, av, false, bv,
                                                        (short)0, acc[nt], false, false);
    }
    __syncthreads();
  }

#pragma unroll
  for (int nt = 0; nt < 4; ++nt) {
#pragma unroll
    for (int i = 0; i < 8; ++i) {
      int r = rbase + (lane >> 4) * 8 + i;
      int c = cbase + nt * 16 + l16;
      int b = r >> 11, n = r & (N_ - 1);
      int h = c >> 6,  d = c & 63;
      size_t idx;
      if (MODE == 1) idx = (((size_t)(b * H_ + h)) * DH_ + d) * (size_t)M_ + n;
      else           idx = (((size_t)(b * H_ + h)) * N_ + n) * (size_t)DH_ + d;
      out[idx] = f2bf(acc[nt][i]);
    }
  }
}

// ---------------------------------------------------------------------------
// Kernel 2: flash attention. grid 512 (= B*H * N/128), block 256 (8 waves).
// Block covers 128 queries of one (b,h); all waves share K/V tiles staged in
// LDS by async global->LDS copies (double buffered, 64 keys per step).
// ---------------------------------------------------------------------------
__global__ __launch_bounds__(256) void attn_kernel(const unsigned short* __restrict__ Q,
                                                   const unsigned short* __restrict__ K,
                                                   const unsigned short* __restrict__ Vt,
                                                   unsigned short* __restrict__ Aout) {
  // pitch 72 shorts = 144 B: 16B-aligned rows, conflict-free b128 reads
  __shared__ unsigned short ldsK[2 * 64 * 72];   // [buf][key row 0..63][d 0..63 + pad]
  __shared__ unsigned short ldsV[2 * 64 * 72];   // [buf][d row 0..63][j 0..63 + pad]
  __shared__ unsigned short ldsP[8 * 16 * 72];   // per-wave P tile 16x64 + pad

  const int tid  = threadIdx.x;
  const int lane = tid & 31;
  const int wave = tid >> 5;
  const int bh   = blockIdx.x >> 4;                       // b*H + h
  const int qb   = (blockIdx.x & 15) * 128 + wave * 16;   // query row base in N
  const int b    = bh >> 3, h = bh & 7;
  const int l16  = lane & 15;
  const int kb8  = (lane >> 4) * 8;
  const int kb16 = (lane >> 4) * 16;

  unsigned short* myP = &ldsP[wave * 16 * 72];

  const unsigned short* kp_base = K  + (size_t)bh * M_ * DH_;   // [key][d]
  const unsigned short* vp_base = Vt + (size_t)bh * DH_ * M_;   // [d][j]

  // Q A-fragments pinned in registers (d 0..31 and 32..63)
  const unsigned short* qp = Q + ((size_t)bh * N_ + qb + l16) * DH_;
  v16bf qa0 = asbf(cat8(*(const v8u16*)(qp + kb16),      *(const v8u16*)(qp + kb16 + 8)));
  v16bf qa1 = asbf(cat8(*(const v8u16*)(qp + 32 + kb16), *(const v8u16*)(qp + 40 + kb16)));

  // each thread copies 2x16B of K and 2x16B of V per 64-key tile
  const int srow = tid >> 3, sseg = tid & 7;   // seg ids 0..255; +256 for second half

  auto issue_tile = [&](int jb, int buf) {
#pragma unroll
    for (int s = 0; s < 2; ++s) {
      int row = srow + s * 32;                 // 0..63
      async_b128(lds_off(&ldsK[(buf * 64 + row) * 72 + sseg * 8]),
                 kp_base + (size_t)(jb + row) * DH_ + sseg * 8);
      async_b128(lds_off(&ldsV[(buf * 64 + row) * 72 + sseg * 8]),
                 vp_base + (size_t)row * M_ + jb + sseg * 8);
    }
  };

  float m[8], l[8];
  v8f acc[4];
#pragma unroll
  for (int i = 0; i < 8; ++i) { m[i] = -1e30f; l[i] = 0.f; }
#pragma unroll
  for (int t = 0; t < 4; ++t) acc[t] = (v8f)0.f;

  const float scale = 0.125f;   // 1/sqrt(64)

  issue_tile(0, 0);

  for (int it = 0; it < M_ / 64; ++it) {
    const int cur = it & 1;
    asm volatile("s_wait_asynccnt 0x0" ::: "memory");  // this wave's copies done
    __syncthreads();                                   // everyone's copies done; prev buf free
    if (it + 1 < M_ / 64) issue_tile((it + 1) * 64, cur ^ 1);   // overlap DMA with compute

    const unsigned short* kbuf = &ldsK[cur * 64 * 72];
    const unsigned short* vbuf = &ldsV[cur * 64 * 72];

    // S tiles: 4 x (16 queries x 16 keys), K-dim = d (two 32-chunks)
    v8f s[4];
#pragma unroll
    for (int t = 0; t < 4; ++t) {
      const unsigned short* kp = kbuf + (t * 16 + l16) * 72 + kb16;
      v16bf b0 = asbf(cat8(*(const v8u16*)kp,        *(const v8u16*)(kp + 8)));
      v16bf b1 = asbf(cat8(*(const v8u16*)(kp + 32), *(const v8u16*)(kp + 40)));
      v8f sv = (v8f)0.f;
      sv = __builtin_amdgcn_wmma_f32_16x16x32_bf16(false, qa0, false, b0, (short)0, sv, false, false);
      sv = __builtin_amdgcn_wmma_f32_16x16x32_bf16(false, qa1, false, b1, (short)0, sv, false, false);
      s[t] = sv;
    }

    // online softmax; C-fragment element i = row (lane/16)*8+i, col lane%16
    float alpha[8];
#pragma unroll
    for (int i = 0; i < 8; ++i) {
      float a0 = s[0][i] * scale, a1 = s[1][i] * scale;
      float a2 = s[2][i] * scale, a3 = s[3][i] * scale;
      float t = fmaxf(fmaxf(a0, a1), fmaxf(a2, a3));
      t = fmaxf(t, __shfl_xor(t, 1, 32));
      t = fmaxf(t, __shfl_xor(t, 2, 32));
      t = fmaxf(t, __shfl_xor(t, 4, 32));
      t = fmaxf(t, __shfl_xor(t, 8, 32));
      float nm = fmaxf(m[i], t);
      alpha[i] = __expf(m[i] - nm);
      float p0 = __expf(a0 - nm), p1 = __expf(a1 - nm);
      float p2 = __expf(a2 - nm), p3 = __expf(a3 - nm);
      m[i] = nm;
      float rs = (p0 + p1) + (p2 + p3);
      rs += __shfl_xor(rs, 1, 32);
      rs += __shfl_xor(rs, 2, 32);
      rs += __shfl_xor(rs, 4, 32);
      rs += __shfl_xor(rs, 8, 32);
      l[i] = l[i] * alpha[i] + rs;
      int roff = ((lane >> 4) * 8 + i) * 72;
      myP[roff + l16]      = f2bf(p0);
      myP[roff + 16 + l16] = f2bf(p1);
      myP[roff + 32 + l16] = f2bf(p2);
      myP[roff + 48 + l16] = f2bf(p3);
    }
#pragma unroll
    for (int t = 0; t < 4; ++t)
#pragma unroll
      for (int i = 0; i < 8; ++i) acc[t][i] *= alpha[i];

    asm volatile("s_wait_dscnt 0x0" ::: "memory");   // P visible before A-frag reads

    // P A-fragments (16x32 each, keys 0..31 / 32..63 of this tile)
    const unsigned short* pp = myP + l16 * 72;
    v16bf pa0 = asbf(cat8(*(const v8u16*)(pp + kb16),      *(const v8u16*)(pp + kb16 + 8)));
    v16bf pa1 = asbf(cat8(*(const v8u16*)(pp + 32 + kb16), *(const v8u16*)(pp + 40 + kb16)));

    // acc += P @ V
#pragma unroll
    for (int nt = 0; nt < 4; ++nt) {
      const unsigned short* vp = vbuf + (nt * 16 + l16) * 72;
      v16bf bv0 = asbf(cat8(*(const v8u16*)(vp + kb16),      *(const v8u16*)(vp + kb16 + 8)));
      v16bf bv1 = asbf(cat8(*(const v8u16*)(vp + 32 + kb16), *(const v8u16*)(vp + 40 + kb16)));
      acc[nt] = __builtin_amdgcn_wmma_f32_16x16x32_bf16(false, pa0, false, bv0,
                                                        (short)0, acc[nt], false, false);
      acc[nt] = __builtin_amdgcn_wmma_f32_16x16x32_bf16(false, pa1, false, bv1,
                                                        (short)0, acc[nt], false, false);
    }
  }

  // finalize: divide by row sums, write bf16 [B*N, 512]
  float inv[8];
#pragma unroll
  for (int i = 0; i < 8; ++i) inv[i] = 1.f / l[i];
#pragma unroll
  for (int nt = 0; nt < 4; ++nt)
#pragma unroll
    for (int i = 0; i < 8; ++i) {
      int n = qb + (lane >> 4) * 8 + i;
      int c = h * 64 + nt * 16 + l16;
      Aout[((size_t)(b * N_ + n)) * DIM_ + c] = f2bf(acc[nt][i] * inv[i]);
    }
}

// ---------------------------------------------------------------------------
// Kernel 3: out = A(bf16)[8192,512] @ Wo(f32) + bo -> f32. grid (64,8), block 256.
// ---------------------------------------------------------------------------
__global__ __launch_bounds__(256) void oproj_kernel(const unsigned short* __restrict__ A,
                                                    const float* __restrict__ W,
                                                    const float* __restrict__ bias,
                                                    float* __restrict__ out) {
  __shared__ unsigned short ldsW[64 * 40];
  const int tid  = threadIdx.x;
  const int lane = tid & 31;
  const int wave = tid >> 5;
  const int rbase = blockIdx.x * 128 + wave * 16;
  const int cbase = blockIdx.y * 64;
  const int l16 = lane & 15;
  const int kb8 = (lane >> 4) * 8;

  v8f acc[4];
#pragma unroll
  for (int t = 0; t < 4; ++t) acc[t] = (v8f)0.f;

  const unsigned short* arow = A + (size_t)(rbase + l16) * DIM_;

  for (int kc = 0; kc < DIM_; kc += 32) {
#pragma unroll
    for (int it = 0; it < 8; ++it) {
      int idx = tid + it * 256;
      int k = idx >> 6, n = idx & 63;
      ldsW[n * 40 + k] = f2bf(W[(size_t)(kc + k) * DIM_ + cbase + n]);
    }
    __syncthreads();

    v16bf av = asbf(cat8(*(const v8u16*)(arow + kc + kb8),
                         *(const v8u16*)(arow + kc + 16 + kb8)));
#pragma unroll
    for (int nt = 0; nt < 4; ++nt) {
      const unsigned short* bp = &ldsW[(nt * 16 + l16) * 40 + kb8];
      v16bf bv = asbf(cat8(*(const v8u16*)bp, *(const v8u16*)(bp + 16)));
      acc[nt] = __builtin_amdgcn_wmma_f32_16x16x32_bf16(false, av, false, bv,
                                                        (short)0, acc[nt], false, false);
    }
    __syncthreads();
  }

#pragma unroll
  for (int nt = 0; nt < 4; ++nt)
#pragma unroll
    for (int i = 0; i < 8; ++i) {
      int r = rbase + (lane >> 4) * 8 + i;
      int c = cbase + nt * 16 + l16;
      out[(size_t)r * DIM_ + c] = acc[nt][i] + bias[c];
    }
}

// ---------------------------------------------------------------------------
extern "C" void kernel_launch(void* const* d_in, const int* in_sizes, int n_in,
                              void* d_out, int out_size, void* d_ws, size_t ws_size,
                              hipStream_t stream) {
  const float* x   = (const float*)d_in[0];
  const float* ctx = (const float*)d_in[1];
  const float* Wq  = (const float*)d_in[2];
  const float* Wk  = (const float*)d_in[3];
  const float* Wv  = (const float*)d_in[4];
  const float* Wo  = (const float*)d_in[5];
  const float* bo  = (const float*)d_in[6];
  float* out = (float*)d_out;

  char* ws = (char*)d_ws;
  unsigned short* Qw = (unsigned short*)(ws);                          // 8 MB [B,H,N,64]
  unsigned short* Kw = (unsigned short*)(ws + ((size_t)8  << 20));     // 8 MB [B,H,M,64]
  unsigned short* Vw = (unsigned short*)(ws + ((size_t)16 << 20));     // 8 MB [B,H,64,M]
  unsigned short* Aw = (unsigned short*)(ws + ((size_t)24 << 20));     // 8 MB [B*N,512]

  dim3 g(64, 8), blk(256);
  proj_kernel<0><<<g, blk, 0, stream>>>(x,   Wq, Qw);
  proj_kernel<0><<<g, blk, 0, stream>>>(ctx, Wk, Kw);
  proj_kernel<1><<<g, blk, 0, stream>>>(ctx, Wv, Vw);
  attn_kernel<<<512, blk, 0, stream>>>(Qw, Kw, Vw, Aw);
  oproj_kernel<<<g, blk, 0, stream>>>(Aw, Wo, bo, out);
}
